// LSTM_Siamese_69612829933861
// MI455X (gfx1250) — matrix operand order repaired and checked by reading, not compiled, not run on hardware
//
#include <hip/hip_runtime.h>
#include <hip/hip_bf16.h>
#include <stdint.h>

#define B_   256
#define T_   2048
#define F_   64
#define H1   64
#define H2   32
#define NTHREADS 512

// padded LDS strides (bank-conflict avoidance)
#define XSTR  72    // bf16 elems; 144B rows -> b128 lane reads partition 64 banks
#define H1STR 72
#define H2STR 40
#define G1P   260   // f32; row = 64 cols x 4 gates interleaved, padded
#define G2P   132   // f32; row = 32 cols x 4 gates interleaved, padded

#define LOG2E 1.4426950408889634f

typedef __attribute__((ext_vector_type(16))) __bf16    v16bf;
typedef __attribute__((ext_vector_type(8)))  float     v8f;
typedef __attribute__((ext_vector_type(4)))  float     v4f;
typedef __attribute__((ext_vector_type(4)))  unsigned  v4u;

union Frag {
    v16bf v;
    v4u   q[2];
    unsigned u[8];
};

// hardware f32->bf16 convert (v_cvt_pk_bf16_f32 class)
static __device__ __forceinline__ unsigned short f2bf(float f) {
    __bf16 b = (__bf16)f;
    return __builtin_bit_cast(unsigned short, b);
}
static __device__ __forceinline__ unsigned pack_bf2(float a, float b) {
    return (unsigned)f2bf(a) | ((unsigned)f2bf(b) << 16);
}

static __device__ __forceinline__ float fast_rcp(float x) {
    return __builtin_amdgcn_rcpf(x);    // single v_rcp_f32 (TRANS32)
}
static __device__ __forceinline__ float fast_exp2(float x) {
    return __builtin_amdgcn_exp2f(x);   // v_exp_f32 is exp2 hardware; log2e pre-folded
}
// branch-free gate activation: c0 + c1 * rcp(1 + exp2(c2l*x)), c2l includes log2(e)
//   sigmoid: (0, 1, -log2e)   tanh: (1, -2, 2*log2e)
static __device__ __forceinline__ float gate_act(float x, float c0, float c1, float c2l) {
    return fmaf(c1, fast_rcp(1.0f + fast_exp2(c2l * x)), c0);
}
static __device__ __forceinline__ float fast_tanh(float x) {
    return 1.0f - 2.0f * fast_rcp(1.0f + fast_exp2((2.0f * LOG2E) * x));
}

// low 32 bits of a flat pointer into __shared__ = LDS byte address (addrspace(3) offset)
static __device__ __forceinline__ unsigned lds_addr(const void* p) {
    return (unsigned)(uintptr_t)p;
}

// B-matrix 32x16 bf16 fragment from global f32 weights W [Nrows x Kdim] row-major,
// B = W^T cols [n0, n0+16). Lane l -> col n0+l%16; elem s<8 K=kb+8h+s, s>=8 K=kb+16+8h+(s-8), h=l/16.
static __device__ __forceinline__ v16bf make_bfrag(const float* W, int Kdim,
                                                   int n0, int lane, int kbase) {
    const int n  = lane & 15;
    const int kh = lane >> 4;
    const float* p = W + (size_t)(n0 + n) * Kdim + kbase + kh * 8;
    Frag f;
#pragma unroll
    for (int j = 0; j < 8; ++j) {
        f.u[j] = pack_bf2(p[2 * j + (j >= 4 ? 8 : 0)], p[2 * j + 1 + (j >= 4 ? 8 : 0)]);
    }
    return f.v;
}

// Store activated 16x16 C tile into gate-interleaved layout [row][col][gate].
// C layout: VGPR i, lanes 0-15 -> (M=i, N=lane), lanes 16-31 -> (M=8+i, N=lane-16).
static __device__ __forceinline__ void store_gates(float* gates, int strideP, int gateId,
                                                   int cbase, int lane, v8f acc) {
    const int c  = cbase + (lane & 15);
    const int mb = (lane >> 4) * 8;
#pragma unroll
    for (int i = 0; i < 8; ++i) gates[(mb + i) * strideP + c * 4 + gateId] = acc[i];
}

#define WMMA_BF16(a, b, c) \
    __builtin_amdgcn_wmma_f32_16x16x32_bf16(false, (a), false, (b), (short)0, (c), false, false)

__global__ __launch_bounds__(NTHREADS, 1)
void lstm_siamese_kernel(const float* __restrict__ x1, const float* __restrict__ x2,
                         const float* __restrict__ Wih1, const float* __restrict__ Whh1,
                         const float* __restrict__ bih1, const float* __restrict__ bhh1,
                         const float* __restrict__ Wih2, const float* __restrict__ Whh2,
                         const float* __restrict__ bih2, const float* __restrict__ bhh2,
                         float* __restrict__ out) {
    __shared__ unsigned short xbuf[16 * XSTR];   // x_t tile, bf16 (padded rows)
    __shared__ float          gates1[16 * G1P];  // layer-1 gates, [r][c][ifgo] interleaved
    __shared__ float          c1s[16 * H1];
    __shared__ unsigned short h1s[16 * H1STR];   // layer-1 hidden, bf16 (padded)
    __shared__ float          gates2[16 * G2P];
    __shared__ float          c2s[16 * H2];
    __shared__ unsigned short h2s[16 * H2STR];   // layer-2 hidden, bf16 (padded)

    const int tid  = threadIdx.x;
    const int lane = tid & 31;
    const int wv   = __builtin_amdgcn_readfirstlane(tid >> 5);  // uniform wave id
    const int s    = blockIdx.x >> 4;          // siamese branch: 0 -> x1, 1 -> x2
    const int b0   = (blockIdx.x & 15) * 16;   // batch tile origin
    const float* x = s ? x2 : x1;

    // zero-init recurrent state in LDS (padding included)
    for (int i = tid; i < 16 * H1STR; i += NTHREADS) h1s[i] = 0;
    for (int i = tid; i < 16 * H2STR; i += NTHREADS) h2s[i] = 0;
    for (int i = tid; i < 16 * H1; i += NTHREADS) c1s[i] = 0.0f;
    for (int i = tid; i < 16 * H2; i += NTHREADS) c2s[i] = 0.0f;

    // ---- weights -> per-wave VGPR-resident B fragments (held for entire T loop) ----
    const int n0_1 = wv * 16;  // this wave's 16 gate columns in layer 1
    const v16bf wih1_0 = make_bfrag(Wih1, F_, n0_1, lane, 0);
    const v16bf wih1_1 = make_bfrag(Wih1, F_, n0_1, lane, 32);
    const v16bf whh1_0 = make_bfrag(Whh1, H1, n0_1, lane, 0);
    const v16bf whh1_1 = make_bfrag(Whh1, H1, n0_1, lane, 32);
    const float bias1  = bih1[n0_1 + (lane & 15)] + bhh1[n0_1 + (lane & 15)];
    const int   g1id   = wv >> 2;          // 0:i 1:f 2:g 3:o (uniform)
    const int   c1base = (wv & 3) * 16;    // column offset within the gate
    // branch-free activation constants (wave-uniform, hoisted; log2e folded into c2)
    const float a1c0 = (g1id == 2) ? 1.0f : 0.0f;
    const float a1c1 = (g1id == 2) ? -2.0f : 1.0f;
    const float a1c2 = (g1id == 2) ? (2.0f * LOG2E) : -LOG2E;

    v16bf wih2_0 = {}, wih2_1 = {}, whh2_0 = {};
    float bias2 = 0.0f;
    float a2c0 = 0.0f, a2c1 = 1.0f, a2c2 = -LOG2E;
    int   g2id = 0, c2base = 0;
    if (wv < 8) {
        const int n0_2 = wv * 16;
        wih2_0 = make_bfrag(Wih2, H1, n0_2, lane, 0);
        wih2_1 = make_bfrag(Wih2, H1, n0_2, lane, 32);
        whh2_0 = make_bfrag(Whh2, H2, n0_2, lane, 0);
        bias2  = bih2[n0_2 + (lane & 15)] + bhh2[n0_2 + (lane & 15)];
        g2id   = wv >> 1;
        c2base = (wv & 1) * 16;
        a2c0 = (g2id == 2) ? 1.0f : 0.0f;
        a2c1 = (g2id == 2) ? -2.0f : 1.0f;
        a2c2 = (g2id == 2) ? (2.0f * LOG2E) : -LOG2E;
    }
    __syncthreads();

    // per-wave LDS base addresses for A-fragment loads (hoisted, reused every step)
    const int   fr    = lane & 15;
    const int   fkh   = lane >> 4;
    const unsigned xaddr  = lds_addr(xbuf + fr * XSTR  + fkh * 8);
    const unsigned h1addr = lds_addr(h1s  + fr * H1STR + fkh * 8);
    const unsigned h2addr = lds_addr(h2s  + fr * H2STR + fkh * 8);

    // cooperative x loader mapping: thread -> 2 consecutive features of one tile row;
    // one wave covers one contiguous 256B row of x.
    const int xr = (tid * 2) / F_;
    const int xc = (tid * 2) % F_;
    const float* xrow = x + (size_t)(b0 + xr) * T_ * F_ + xc;

    // elementwise layer-2 updater mapping (also the output mapping)
    const int r2 = tid / H2, col2 = tid % H2;
    float h2_last = 0.0f;

    // software pipeline: x_t held in registers one step ahead of use
    float2 vcur = *(const float2*)xrow;

    for (int t = 0; t < T_; ++t) {
        // ---- phase 0: issue load of x_{t+1}, commit x_t to LDS as bf16 ----
        const int tn = (t + 1 < T_) ? (t + 1) : t;      // clamp (redundant last load)
        const float* pn = xrow + (size_t)tn * F_;
        float2 vnext = *(const float2*)pn;              // waits next iteration, not now
        __builtin_prefetch(pn + F_, 0, 1);              // t+2 -> global_prefetch (speculative)
        *(unsigned*)&xbuf[xr * XSTR + xc] = pack_bf2(vcur.x, vcur.y);
        __syncthreads();

        // ---- phase 1: layer-1 gates. Batch all 8 fragment loads into distinct quad
        //      registers with ONE s_wait_dscnt, then 4 WMMAs over two chains ----
        {
            Frag fx0, fx1, fh0, fh1;
            asm volatile(
                "ds_load_b128 %0, %8\n\t"
                "ds_load_b128 %1, %8 offset:32\n\t"
                "ds_load_b128 %2, %8 offset:64\n\t"
                "ds_load_b128 %3, %8 offset:96\n\t"
                "ds_load_b128 %4, %9\n\t"
                "ds_load_b128 %5, %9 offset:32\n\t"
                "ds_load_b128 %6, %9 offset:64\n\t"
                "ds_load_b128 %7, %9 offset:96\n\t"
                "s_wait_dscnt 0x0"
                : "=v"(fx0.q[0]), "=v"(fx0.q[1]), "=v"(fx1.q[0]), "=v"(fx1.q[1]),
                  "=v"(fh0.q[0]), "=v"(fh0.q[1]), "=v"(fh1.q[0]), "=v"(fh1.q[1])
                : "v"(xaddr), "v"(h1addr)
                : "memory");
            v8f accA, accB;
#pragma unroll
            for (int i = 0; i < 8; ++i) { accA[i] = bias1; accB[i] = 0.0f; }
            accA = WMMA_BF16(fx0.v, wih1_0, accA);
            accB = WMMA_BF16(fx1.v, wih1_1, accB);
            accA = WMMA_BF16(fh0.v, whh1_0, accA);
            accB = WMMA_BF16(fh1.v, whh1_1, accB);
#pragma unroll
            for (int i = 0; i < 8; ++i)
                accA[i] = gate_act(accA[i] + accB[i], a1c0, a1c1, a1c2);
            store_gates(gates1, G1P, g1id, c1base, lane, accA);
        }
        __syncthreads();

        // ---- phase 2: c1/h1 elementwise update; one b128 gate read per element ----
#pragma unroll
        for (int k = 0; k < 2; ++k) {
            const int e = tid + k * NTHREADS;
            const int r = e / H1, c = e % H1;
            const v4f g4 = *(const v4f*)&gates1[r * G1P + c * 4];   // i,f,g,o
            const float cn = g4.y * c1s[e] + g4.x * g4.z;
            c1s[e] = cn;
            h1s[r * H1STR + c] = f2bf(g4.w * fast_tanh(cn));
        }
        __syncthreads();

        // ---- phase 3: layer-2 gates, waves 0..7; 6 batched loads, one wait, 3 WMMAs ----
        if (wv < 8) {
            Frag fa0, fa1, fah;
            asm volatile(
                "ds_load_b128 %0, %6\n\t"
                "ds_load_b128 %1, %6 offset:32\n\t"
                "ds_load_b128 %2, %6 offset:64\n\t"
                "ds_load_b128 %3, %6 offset:96\n\t"
                "ds_load_b128 %4, %7\n\t"
                "ds_load_b128 %5, %7 offset:32\n\t"
                "s_wait_dscnt 0x0"
                : "=v"(fa0.q[0]), "=v"(fa0.q[1]), "=v"(fa1.q[0]), "=v"(fa1.q[1]),
                  "=v"(fah.q[0]), "=v"(fah.q[1])
                : "v"(h1addr), "v"(h2addr)
                : "memory");
            v8f accA, accB;
#pragma unroll
            for (int i = 0; i < 8; ++i) { accA[i] = bias2; accB[i] = 0.0f; }
            accA = WMMA_BF16(fa0.v, wih2_0, accA);
            accB = WMMA_BF16(fa1.v, wih2_1, accB);
            accA = WMMA_BF16(fah.v, whh2_0, accA);
#pragma unroll
            for (int i = 0; i < 8; ++i)
                accA[i] = gate_act(accA[i] + accB[i], a2c0, a2c1, a2c2);
            store_gates(gates2, G2P, g2id, c2base, lane, accA);
        }
        __syncthreads();

        // ---- phase 4: c2/h2 elementwise update (512 elems, 1 per thread) ----
        // No trailing barrier: h2s/c2s written here are next read in phase 3 of the
        // following iteration, separated by >=2 barriers.
        {
            const v4f g4 = *(const v4f*)&gates2[r2 * G2P + col2 * 4];  // i,f,g,o
            const float cn = g4.y * c2s[tid] + g4.x * g4.z;
            c2s[tid] = cn;
            h2_last  = g4.w * fast_tanh(cn);
            h2s[r2 * H2STR + col2] = f2bf(h2_last);
        }

        vcur = vnext;
    }

    // final hidden of layer 2 -> out; thread's (r2,col2) mapping matches output layout
    out[(size_t)s * B_ * H2 + (size_t)(b0 + r2) * H2 + col2] = h2_last;
}

extern "C" void kernel_launch(void* const* d_in, const int* in_sizes, int n_in,
                              void* d_out, int out_size, void* d_ws, size_t ws_size,
                              hipStream_t stream) {
    (void)in_sizes; (void)n_in; (void)out_size; (void)d_ws; (void)ws_size;
    const float* x1   = (const float*)d_in[0];
    const float* x2   = (const float*)d_in[1];
    const float* Wih1 = (const float*)d_in[2];
    const float* Whh1 = (const float*)d_in[3];
    const float* bih1 = (const float*)d_in[4];
    const float* bhh1 = (const float*)d_in[5];
    const float* Wih2 = (const float*)d_in[6];
    const float* Whh2 = (const float*)d_in[7];
    const float* bih2 = (const float*)d_in[8];
    const float* bhh2 = (const float*)d_in[9];
    float* out = (float*)d_out;

    // 2 siamese branches x 16 batch tiles of M=16; 512 threads = 16 wave32s per WG
    lstm_siamese_kernel<<<dim3(32), dim3(NTHREADS), 0, stream>>>(
        x1, x2, Wih1, Whh1, bih1, bhh1, Wih2, Whh2, bih2, bhh2, out);
}
